// backbone_pointnet2_48369921687902
// MI455X (gfx1250) — matrix operand order repaired and checked by reading, not compiled, not run on hardware
//
#include <hip/hip_runtime.h>
#include <stdint.h>

// ---------- CDNA5 WMMA types ----------
typedef __attribute__((ext_vector_type(16))) __bf16 v16bf;
typedef __attribute__((ext_vector_type(8)))  float  v8f;

union BF16Frag { v16bf v; uint16_t u[16]; uint32_t d[8]; };

__device__ __forceinline__ float bf2f(uint16_t h){
  union { uint32_t u; float f; } c; c.u = ((uint32_t)h) << 16; return c.f;
}
__device__ __forceinline__ uint16_t f2bf(float f){
  union { float f; uint32_t u; } c; c.f = f;
  uint32_t u = c.u;
  u += 0x7FFFu + ((u >> 16) & 1u);      // round to nearest even
  return (uint16_t)(u >> 16);
}

// ---------- vk2 layout: channel-pair interleaved activations ----------
// bf16 tensor (B,C,M): element (c,m) at u16 index (c>>1)*2M + 2m + (c&1).
// K-pairs are dword-contiguous => WMMA B-fragments load as 8x ds_load_b32,
// and pair-rows are m-contiguous => async b128 DMA stages them directly.
__device__ __forceinline__ size_t vk2(int c, int m, int M){
  return (size_t)(c >> 1)*((size_t)M*2) + (size_t)m*2 + (size_t)(c & 1);
}
__device__ __forceinline__ size_t vk2base(int b, int C, int M){
  return (size_t)b * (size_t)((C + 1) >> 1) * ((size_t)M*2);
}

// ---------- CDNA5 async global->LDS (ASYNCcnt-tracked DMA, no VGPR data) ----------
__device__ __forceinline__ void async_g2l_b128(uint32_t lds_byte_off, const void* gaddr){
  asm volatile("global_load_async_to_lds_b128 %0, %1, off"
               :: "v"(lds_byte_off), "v"(gaddr) : "memory");
}
#define WAIT_ASYNC() asm volatile("s_wait_asynccnt 0x0" ::: "memory")

// =====================================================================
// GEMM + bias + BN + activation.  Y[b,o,m] = act(BN(sum_c W[o,c]*X[b,c,m]+b))
// X: (B,Cin,M) bf16 in vk2 layout.  Tile: 16 (O) x 128 (M) per block;
// 8 waves, one v_wmma_f32_16x16x32_bf16 per wave per 32-deep K-chunk.
// X tiles staged with global_load_async_to_lds_b128, double-buffered so
// the async DMA of chunk k+1 overlaps the WMMA of chunk k.
// =====================================================================
__global__ void __launch_bounds__(256)
gemm_bn_act_kernel(const uint16_t* __restrict__ X, uint16_t* __restrict__ Ybf,
                   float* __restrict__ Yf,
                   const float* __restrict__ W, const float* __restrict__ bias,
                   const float* __restrict__ gam, const float* __restrict__ bet,
                   const float* __restrict__ mean, const float* __restrict__ var,
                   int Cin, int Cout, int M, int act)
{
  __shared__ uint16_t Wl[2][16*32];
  __shared__ uint32_t Xl[2][16*132];   // 16 K-pair rows x 128 dwords (+4 pad, rows 16B aligned)
  __shared__ float sArr[16], tArr[16];

  const int tid  = threadIdx.x;
  const int lane = tid & 31, wave = tid >> 5;
  const int o0 = blockIdx.y * 16;
  const int m0 = blockIdx.x * 128;
  const int b  = blockIdx.z;
  const int P  = (Cin + 1) >> 1;                        // K-pair rows in X
  const uint32_t* X32 = (const uint32_t*)X;
  const size_t x32base = (size_t)b * (size_t)P * (size_t)M;  // dword units

  if (tid < 16){                       // fused epilogue scale/shift per out-ch
    int o = o0 + tid;
    float s = 1.f, t = 0.f;
    if (o < Cout){
      if (gam){ float rs = rsqrtf(var[o] + 1e-5f); s = gam[o]*rs;
                t = (bias[o] - mean[o])*s + bet[o]; }
      else    { t = bias[o]; }
    }
    sArr[tid] = s; tArr[tid] = t;
  }
  __builtin_prefetch(W + (size_t)(o0 + (tid & 15))*Cin, 0, 0);   // warm L2 (global_prefetch_b8)

  auto stageW = [&](int kc, int buf){  // 16x32 f32 -> bf16, zero-pad ragged edges
    for (int t2 = tid; t2 < 16*32; t2 += 256){
      int r = t2 >> 5, cc = t2 & 31;
      int o = o0 + r, c = kc + cc;
      float w = (o < Cout && c < Cin) ? W[(size_t)o*Cin + c] : 0.f;
      Wl[buf][t2] = f2bf(w);
    }
  };
  auto stageX = [&](int kc, int buf){  // 16 pair-rows x 128 dwords
    if (kc + 32 <= Cin){
      // full K-chunk: async DMA, 512 x 16B segments (2 per thread).
      // Over-read columns (only when M==64) land in masked-out output
      // columns and stay inside the workspace allocation.
      uint32_t lds0 = (uint32_t)(uintptr_t)&Xl[buf][0];
      for (int sg = tid; sg < 512; sg += 256){
        int p  = sg >> 5;              // 32 segments per 128-dword row
        int m4 = (sg & 31) << 2;       // dword offset within row
        async_g2l_b128(lds0 + (uint32_t)(p*132 + m4)*4u,
                       (const void*)(X32 + x32base + (size_t)((kc >> 1) + p)*M + m0 + m4));
      }
    } else {
      // ragged Cin tail: guarded synchronous staging with zero-fill
      for (int t2 = tid; t2 < 16*128; t2 += 256){
        int p = t2 >> 7, m = t2 & 127;
        int c0 = kc + 2*p, mg = m0 + m;
        uint32_t vv = 0;
        if (c0 < Cin && mg < M){
          vv = X32[x32base + (size_t)((kc >> 1) + p)*M + mg];
          if (c0 + 1 >= Cin) vv &= 0xFFFFu;     // odd-channel tail never contaminates
        }
        Xl[buf][p*132 + m] = vv;
      }
    }
  };

  const int nk = (Cin + 31) >> 5;
  stageW(0, 0); stageX(0, 0);
  WAIT_ASYNC();
  __syncthreads();

  v8f acc = {};
  const int ln = lane & 15, hlf = lane >> 4;
  for (int ck = 0; ck < nk; ++ck){
    const int buf = ck & 1;
    if (ck + 1 < nk){ stageW((ck + 1)*32, buf ^ 1); stageX((ck + 1)*32, buf ^ 1); }

    BF16Frag Af, Bf;
    #pragma unroll
    for (int j = 0; j < 16; ++j){
      int v = j >> 1;
      // A (16x32 bf16): lane row = ln, K = 16*hlf + j (contiguous, matches B)
      int ka = 16*hlf + j;
      Af.u[j] = Wl[buf][ln*32 + ka];
    }
    #pragma unroll
    for (int p = 0; p < 8; ++p)
      // B (32x16 bf16): lane col = ln, K-pair p+8*hlf -> one dword
      Bf.d[p] = Xl[buf][(8*hlf + p)*132 + wave*16 + ln];

    acc = __builtin_amdgcn_wmma_f32_16x16x32_bf16(false, Af.v, false, Bf.v,
                                                  (short)0, acc, false, false);
    WAIT_ASYNC();          // own async stores for next chunk complete
    __syncthreads();       // publish LDS to all waves / protect reused buffer
  }

  const int colm = m0 + wave*16 + (lane & 15);
  float yv[8];
  #pragma unroll
  for (int i = 0; i < 8; ++i){
    int r = i + 8*hlf;                 // D row = O index within tile
    float y = acc[i] * sArr[r] + tArr[r];
    if (act == 1)      y = fmaxf(y, 0.f);
    else if (act == 2) y = (y >= 0.f) ? y : 0.2f*y;
    yv[i] = y;
  }
  if (Ybf && colm < M){                // vk2 out: pack channel pairs -> b32 stores
    uint32_t* Y32 = (uint32_t*)Ybf;
    const int Po = (Cout + 1) >> 1;
    #pragma unroll
    for (int i = 0; i < 8; i += 2){
      int o = o0 + i + 8*hlf;
      if (o + 1 < Cout)
        Y32[(size_t)b*Po*M + (size_t)(o >> 1)*M + colm] =
            (uint32_t)f2bf(yv[i]) | ((uint32_t)f2bf(yv[i+1]) << 16);
    }
  }
  if (Yf && colm < M){                 // plain f32 out (head logits)
    #pragma unroll
    for (int i = 0; i < 8; ++i){
      int o = o0 + i + 8*hlf;
      if (o < Cout) Yf[((size_t)b*Cout + o)*(size_t)M + colm] = yv[i];
    }
  }
}

// =====================================================================
// Farthest point sampling: one block per batch, iterative argmax.
// =====================================================================
__global__ void __launch_bounds__(256)
fps_kernel(const float* __restrict__ xyz, int N, int npoint, float* __restrict__ new_xyz)
{
  __shared__ float dist[4096];
  __shared__ float rv[256];
  __shared__ int   ri[256];
  __shared__ int   sel[1024];
  const int b = blockIdx.x, tid = threadIdx.x;
  const float* xb = xyz + (size_t)b*N*3;
  for (int i = tid; i < N; i += 256) dist[i] = 1e10f;
  __syncthreads();
  int far = 0;
  for (int it = 0; it < npoint; ++it){
    if (tid == 0) sel[it] = far;
    float fx = xb[far*3+0], fy = xb[far*3+1], fz = xb[far*3+2];
    float best = -1.f; int bi = 0;
    for (int i = tid; i < N; i += 256){
      float dx = xb[i*3+0]-fx, dy = xb[i*3+1]-fy, dz = xb[i*3+2]-fz;
      float d = dx*dx + dy*dy + dz*dz;
      float dd = fminf(dist[i], d);
      dist[i] = dd;
      if (dd > best){ best = dd; bi = i; }
    }
    rv[tid] = best; ri[tid] = bi;
    __syncthreads();
    for (int s = 128; s > 0; s >>= 1){
      if (tid < s){
        float v2 = rv[tid+s]; int i2 = ri[tid+s];
        if (v2 > rv[tid] || (v2 == rv[tid] && i2 < ri[tid])){ rv[tid]=v2; ri[tid]=i2; }
      }
      __syncthreads();
    }
    far = ri[0];
    __syncthreads();
  }
  for (int s = tid; s < npoint; s += 256){
    int id = sel[s];
    new_xyz[((size_t)b*npoint + s)*3 + 0] = xb[id*3+0];
    new_xyz[((size_t)b*npoint + s)*3 + 1] = xb[id*3+1];
    new_xyz[((size_t)b*npoint + s)*3 + 2] = xb[id*3+2];
  }
}

// =====================================================================
// Ball query + gather (fused). One wave32 per query center; ordered
// stream compaction of first-K in-radius indices via ballot.
// =====================================================================
__global__ void __launch_bounds__(256)
ballquery_group_kernel(const float* __restrict__ xyz, const float* __restrict__ ctr,
                       const void* __restrict__ feats, int feats_f32,
                       uint16_t* __restrict__ grouped,
                       int N, int S, int C, int K, float r2)
{
  __shared__ int idxs[8][128];
  const int lane = threadIdx.x & 31, wave = threadIdx.x >> 5;
  const int s = blockIdx.x * 8 + wave;
  const int b = blockIdx.y;
  int* my = idxs[wave];

  float cx = 0.f, cy = 0.f, cz = 0.f;
  if (s < S){
    cx = ctr[((size_t)b*S + s)*3 + 0];
    cy = ctr[((size_t)b*S + s)*3 + 1];
    cz = ctr[((size_t)b*S + s)*3 + 2];
  }
  int cnt = 0;
  for (int n0 = 0; n0 < N && cnt < K; n0 += 32){
    int n = n0 + lane;
    bool ok = false;
    if (s < S && n < N){
      float dx = xyz[((size_t)b*N + n)*3 + 0] - cx;
      float dy = xyz[((size_t)b*N + n)*3 + 1] - cy;
      float dz = xyz[((size_t)b*N + n)*3 + 2] - cz;
      ok = (dx*dx + dy*dy + dz*dz) < r2;
    }
    unsigned mask = (unsigned)__ballot(ok);           // wave32: low 32 bits
    int pos = cnt + __popc(mask & ((1u << lane) - 1u));
    if (ok && pos < K) my[pos] = n;
    cnt += __popc(mask);
  }
  if (cnt > K) cnt = K;
  __syncthreads();
  int first = (cnt > 0) ? my[0] : 0;                  // pad rule of reference
  for (int k = cnt + lane; k < K; k += 32) my[k] = first;
  __syncthreads();

  if (s < S){
    const int SK = S*K;
    const size_t gb = vk2base(b, C, SK);
    if (feats_f32){
      const float* F = (const float*)feats;
      for (int c = 0; c < C; ++c){
        const float* rowp = F + ((size_t)b*C + c)*N;
        for (int k = lane; k < K; k += 32)
          grouped[gb + vk2(c, s*K + k, SK)] = f2bf(rowp[my[k]]);
      }
    } else {   // src feats are vk2 (B,C,N)
      const uint16_t* F = (const uint16_t*)feats;
      for (int c = 0; c < C; ++c){
        const size_t fb = vk2base(b, C, N);
        for (int k = lane; k < K; k += 32)
          grouped[gb + vk2(c, s*K + k, SK)] = F[fb + vk2(c, my[k], N)];
      }
    }
  }
}

// ---------- max over K: (B,C,S*K) -> (B,C,S) (vk2), optional plain f32 copy ----------
__global__ void maxpool_kernel(const uint16_t* __restrict__ X, uint16_t* __restrict__ Obf,
                               float* __restrict__ Of, int C, int S, int K)
{
  int t = blockIdx.x*256 + threadIdx.x, b = blockIdx.y;
  if (t >= C*S) return;
  int c = t / S, s2 = t % S;
  const uint16_t* xb = X + vk2base(b, C, S*K);
  float m = -3e38f;
  for (int k = 0; k < K; ++k) m = fmaxf(m, bf2f(xb[vk2(c, s2*K + k, S*K)]));
  Obf[vk2base(b, C, S) + vk2(c, s2, S)] = f2bf(m);
  if (Of) Of[((size_t)b*C + c)*S + s2] = m;
}

// ---------- 3-NN for feature propagation ----------
__global__ void knn3_kernel(const float* __restrict__ unk, const float* __restrict__ kn,
                            int Nu, int S, int* __restrict__ idx3, float* __restrict__ w3)
{
  int n = blockIdx.x*256 + threadIdx.x, b = blockIdx.y;
  if (n >= Nu) return;
  float ux = unk[((size_t)b*Nu+n)*3+0], uy = unk[((size_t)b*Nu+n)*3+1], uz = unk[((size_t)b*Nu+n)*3+2];
  float d0 = 3e38f, d1 = 3e38f, d2 = 3e38f; int i0 = 0, i1 = 0, i2 = 0;
  for (int s2 = 0; s2 < S; ++s2){
    float dx = kn[((size_t)b*S+s2)*3+0]-ux, dy = kn[((size_t)b*S+s2)*3+1]-uy, dz = kn[((size_t)b*S+s2)*3+2]-uz;
    float d = dx*dx + dy*dy + dz*dz;
    if (d < d0){ d2=d1;i2=i1; d1=d0;i1=i0; d0=d;i0=s2; }
    else if (d < d1){ d2=d1;i2=i1; d1=d;i1=s2; }
    else if (d < d2){ d2=d;i2=s2; }
  }
  float w0 = 1.f/(d0+1e-8f), w1 = 1.f/(d1+1e-8f), w2 = 1.f/(d2+1e-8f);
  float ws = w0 + w1 + w2;
  size_t o = ((size_t)b*Nu + n)*3;
  idx3[o]=i0; idx3[o+1]=i1; idx3[o+2]=i2;
  w3[o]=w0/ws; w3[o+1]=w1/ws; w3[o+2]=w2/ws;
}

// ---------- weighted 3-NN interpolation into concat buffer (ch 0..Ck) ----------
__global__ void interp_kernel(const uint16_t* __restrict__ kf, const int* __restrict__ idx3,
                              const float* __restrict__ w3, uint16_t* __restrict__ out,
                              int Ck, int Ctot, int S, int Nu)
{
  int t = blockIdx.x*256 + threadIdx.x, b = blockIdx.y;
  if (t >= Ck*Nu) return;
  int c = t / Nu, n = t % Nu;
  size_t o3 = ((size_t)b*Nu + n)*3;
  const uint16_t* kb = kf + vk2base(b, Ck, S);
  float v = w3[o3+0]*bf2f(kb[vk2(c, idx3[o3+0], S)])
          + w3[o3+1]*bf2f(kb[vk2(c, idx3[o3+1], S)])
          + w3[o3+2]*bf2f(kb[vk2(c, idx3[o3+2], S)]);
  out[vk2base(b, Ctot, Nu) + vk2(c, n, Nu)] = f2bf(v);
}

// ---------- copy bf16 feats into concat buffer at channel offset ----------
__global__ void copych_kernel(const uint16_t* __restrict__ src, uint16_t* __restrict__ out,
                              int Cs, int Ctot, int cofs, int Nu)
{
  int t = blockIdx.x*256 + threadIdx.x, b = blockIdx.y;
  if (t >= Cs*Nu) return;
  int c = t / Nu, n = t % Nu;
  out[vk2base(b, Ctot, Nu) + vk2(cofs + c, n, Nu)] = src[vk2base(b, Cs, Nu) + vk2(c, n, Nu)];
}

// ---------- broadcast global feature over points (fp4 interp part) ----------
__global__ void bcast_kernel(const uint16_t* __restrict__ g, uint16_t* __restrict__ out,
                             int Ck, int Ctot, int Nu)
{
  int t = blockIdx.x*256 + threadIdx.x, b = blockIdx.y;
  if (t >= Ck*Nu) return;
  int c = t / Nu, n = t % Nu;
  // l4f is (B,512,1) in vk2 == plain (B,512)
  out[vk2base(b, Ctot, Nu) + vk2(c, n, Nu)] = g[(size_t)b*Ck + c];
}

// ---------- fp1 unknown feats: [xyz^T (3ch); points (6ch)] -> concat ----------
__global__ void fp1unk_kernel(const float* __restrict__ xyz, const float* __restrict__ pts,
                              uint16_t* __restrict__ out, int N, int Ctot, int cofs)
{
  int t = blockIdx.x*256 + threadIdx.x, b = blockIdx.y;
  if (t >= 9*N) return;
  int c = t / N, n = t % N;
  float v = (c < 3) ? xyz[((size_t)b*N + n)*3 + c]
                    : pts[((size_t)b*6 + (c-3))*N + n];
  out[vk2base(b, Ctot, N) + vk2(cofs + c, n, N)] = f2bf(v);
}

// ---------- point_features: (B,128,N) vk2 bf16 -> (B,N,128) f32 ----------
__global__ void transpose_pf_kernel(const uint16_t* __restrict__ l0, float* __restrict__ out,
                                    int C, int N)
{
  int t = blockIdx.x*256 + threadIdx.x, b = blockIdx.y;
  if (t >= C*N) return;
  int c = t / N, n = t % N;
  out[((size_t)b*N + n)*C + c] = bf2f(l0[vk2base(b, C, N) + vk2(c, n, N)]);
}

// ---------- softmax over N axis of sem4 (strided slices of (13,N) logits) ----------
__global__ void __launch_bounds__(256)
softmax_kernel(const float* __restrict__ sem, float* __restrict__ y, int N, int Kc)
{
  __shared__ float red[256];
  int b = blockIdx.x / Kc, k = blockIdx.x % Kc;
  int tid = threadIdx.x;
  const float* base = sem + (size_t)b*Kc*N;   // sem4[b,n,k] == base[n*Kc + k]
  float mx = -3e38f;
  for (int n = tid; n < N; n += 256) mx = fmaxf(mx, base[(size_t)n*Kc + k]);
  red[tid] = mx; __syncthreads();
  for (int s = 128; s > 0; s >>= 1){ if (tid < s) red[tid] = fmaxf(red[tid], red[tid+s]); __syncthreads(); }
  mx = red[0]; __syncthreads();
  float sum = 0.f;
  for (int n = tid; n < N; n += 256) sum += __expf(base[(size_t)n*Kc + k] - mx);
  red[tid] = sum; __syncthreads();
  for (int s = 128; s > 0; s >>= 1){ if (tid < s) red[tid] += red[tid+s]; __syncthreads(); }
  sum = red[0];
  float inv = 1.f / sum;
  for (int n = tid; n < N; n += 256)
    y[(size_t)b*N*Kc + (size_t)n*Kc + k] = __expf(base[(size_t)n*Kc + k] - mx) * inv;
}

// =====================================================================
// Host orchestration
// =====================================================================
extern "C" void kernel_launch(void* const* d_in, const int* in_sizes, int n_in,
                              void* d_out, int out_size, void* d_ws, size_t ws_size,
                              hipStream_t stream)
{
  (void)in_sizes; (void)n_in; (void)out_size; (void)ws_size;
  const int B = 16, N0 = 4096;
  const float* xyz = (const float*)d_in[0];
  const float* pts = (const float*)d_in[1];

  struct Lyr { const float *W,*b,*g,*be,*mn,*vr; int cin,cout; };
  Lyr Ls[25]; int nl = 0;
  int pi = 2;
  auto take = [&](void)->const float*{ return (const float*)d_in[pi++]; };
  const int specs[8][5] = {
    {6,32,32,64,0},{64,64,64,128,0},{128,128,128,256,0},{256,256,256,512,0},
    {768,256,256,0,0},{384,256,256,0,0},{320,256,128,0,0},{137,128,128,128,128}};
  const int lcnt[8] = {3,3,3,3,2,2,2,4};
  for (int gi = 0; gi < 8; ++gi)
    for (int l = 0; l < lcnt[gi]; ++l){
      Lyr L; L.cin = specs[gi][l]; L.cout = specs[gi][l+1];
      L.W = take(); L.b = take(); L.g = take(); L.be = take(); L.mn = take(); L.vr = take();
      Ls[nl++] = L;
    }
  const int hspec[4] = {128,128,64,13};
  for (int l = 0; l < 3; ++l){
    Lyr L; L.cin = hspec[l]; L.cout = hspec[l+1];
    L.W = take(); L.b = take(); L.g = L.be = L.mn = L.vr = nullptr;
    Ls[nl++] = L;
  }
  // layer indices: sa1:0-2 sa2:3-5 sa3:6-8 sa4:9-11 fp4:12-13 fp3:14-15 fp2:16-17 fp1:18-21 head:22-24

  // ---- workspace carve ----
  char* wsb = (char*)d_ws;
  size_t off = 0;
  auto carve = [&](size_t bytes)->char*{
    char* p = wsb + off; off = (off + bytes + 255) & ~(size_t)255; return p;
  };
  const size_t PING = (size_t)B * 2097152;           // max padded C*M elems per batch
  uint16_t* pA  = (uint16_t*)carve(PING*2);
  uint16_t* pB  = (uint16_t*)carve(PING*2);
  float* l1x = (float*)carve((size_t)B*1024*3*4);
  float* l2x = (float*)carve((size_t)B*256*3*4);
  float* l3x = (float*)carve((size_t)B*64*3*4);
  uint16_t* l1f = (uint16_t*)carve((size_t)B*64*1024*2);
  uint16_t* l2f = (uint16_t*)carve((size_t)B*128*256*2);
  uint16_t* l3f = (uint16_t*)carve((size_t)B*256*64*2);
  uint16_t* l4f = (uint16_t*)carve((size_t)B*512*2);
  uint16_t* f3  = (uint16_t*)carve((size_t)B*256*64*2);
  uint16_t* f2b = (uint16_t*)carve((size_t)B*256*256*2);
  uint16_t* f1b = (uint16_t*)carve((size_t)B*128*1024*2);
  uint16_t* l0  = (uint16_t*)carve((size_t)B*128*4096*2);
  int*   idx3 = (int*)carve((size_t)B*4096*3*4);
  float* w3   = (float*)carve((size_t)B*4096*3*4);

  // ---- output regions (concatenated flat, reference return order) ----
  float* OUT   = (float*)d_out;
  float* o_pf  = OUT;                          // (B,4096,128)
  float* o_gf  = OUT + (size_t)B*4096*128;     // (B,512)
  float* o_ys  = o_gf + (size_t)B*512;         // (B,4096,13)
  float* o_sem = o_ys + (size_t)B*4096*13;     // sem4 == (B,13,4096) flat

  auto gemm = [&](const uint16_t* Xp, uint16_t* Yb, float* Yfp, const Lyr& L, int M, int act){
    dim3 gr((unsigned)((M + 127)/128), (unsigned)((L.cout + 15)/16), (unsigned)B);
    gemm_bn_act_kernel<<<gr, 256, 0, stream>>>(Xp, Yb, Yfp, L.W, L.b, L.g, L.be, L.mn, L.vr,
                                               L.cin, L.cout, M, act);
  };
  auto e1 = [&](int cnt){ return dim3((unsigned)((cnt + 255)/256), (unsigned)B); };

  // ---- SA1: 4096 -> 1024, r=0.1, K=32, MLP 6-32-32-64 ----
  fps_kernel<<<B, 256, 0, stream>>>(xyz, N0, 1024, l1x);
  ballquery_group_kernel<<<dim3(1024/8, B), 256, 0, stream>>>(xyz, l1x, pts, 1, pA, N0, 1024, 6, 32, 0.01f);
  gemm(pA, pB, nullptr, Ls[0], 32768, 1);
  gemm(pB, pA, nullptr, Ls[1], 32768, 1);
  gemm(pA, pB, nullptr, Ls[2], 32768, 1);
  maxpool_kernel<<<e1(64*1024), 256, 0, stream>>>(pB, l1f, nullptr, 64, 1024, 32);

  // ---- SA2: 1024 -> 256, r=0.2, K=64, MLP 64-64-64-128 ----
  fps_kernel<<<B, 256, 0, stream>>>(l1x, 1024, 256, l2x);
  ballquery_group_kernel<<<dim3(256/8, B), 256, 0, stream>>>(l1x, l2x, l1f, 0, pA, 1024, 256, 64, 64, 0.04f);
  gemm(pA, pB, nullptr, Ls[3], 16384, 1);
  gemm(pB, pA, nullptr, Ls[4], 16384, 1);
  gemm(pA, pB, nullptr, Ls[5], 16384, 1);
  maxpool_kernel<<<e1(128*256), 256, 0, stream>>>(pB, l2f, nullptr, 128, 256, 64);

  // ---- SA3: 256 -> 64, r=0.4, K=128, MLP 128-128-128-256 ----
  fps_kernel<<<B, 256, 0, stream>>>(l2x, 256, 64, l3x);
  ballquery_group_kernel<<<dim3(64/8, B), 256, 0, stream>>>(l2x, l3x, l2f, 0, pA, 256, 64, 128, 128, 0.16f);
  gemm(pA, pB, nullptr, Ls[6], 8192, 1);
  gemm(pB, pA, nullptr, Ls[7], 8192, 1);
  gemm(pA, pB, nullptr, Ls[8], 8192, 1);
  maxpool_kernel<<<e1(256*64), 256, 0, stream>>>(pB, l3f, nullptr, 256, 64, 128);

  // ---- SA4: global MLP 256-256-256-512 over 64 points, max -> (B,512) ----
  gemm(l3f, pA, nullptr, Ls[9],  64, 1);
  gemm(pA,  pB, nullptr, Ls[10], 64, 1);
  gemm(pB,  pA, nullptr, Ls[11], 64, 1);
  maxpool_kernel<<<e1(512), 256, 0, stream>>>(pA, l4f, o_gf, 512, 1, 64);

  // ---- FP4: broadcast l4 + l3 -> MLP 768-256-256 ----
  bcast_kernel<<<e1(512*64), 256, 0, stream>>>(l4f, pA, 512, 768, 64);
  copych_kernel<<<e1(256*64), 256, 0, stream>>>(l3f, pA, 256, 768, 512, 64);
  gemm(pA, pB, nullptr, Ls[12], 64, 1);
  gemm(pB, f3, nullptr, Ls[13], 64, 1);

  // ---- FP3: interp(l2x<-l3x, f3) + l2 -> MLP 384-256-256 ----
  knn3_kernel<<<e1(256), 256, 0, stream>>>(l2x, l3x, 256, 64, idx3, w3);
  interp_kernel<<<e1(256*256), 256, 0, stream>>>(f3, idx3, w3, pA, 256, 384, 64, 256);
  copych_kernel<<<e1(128*256), 256, 0, stream>>>(l2f, pA, 128, 384, 256, 256);
  gemm(pA, pB, nullptr, Ls[14], 256, 1);
  gemm(pB, f2b, nullptr, Ls[15], 256, 1);

  // ---- FP2: interp(l1x<-l2x, f2) + l1 -> MLP 320-256-128 ----
  knn3_kernel<<<e1(1024), 256, 0, stream>>>(l1x, l2x, 1024, 256, idx3, w3);
  interp_kernel<<<e1(256*1024), 256, 0, stream>>>(f2b, idx3, w3, pA, 256, 320, 256, 1024);
  copych_kernel<<<e1(64*1024), 256, 0, stream>>>(l1f, pA, 64, 320, 256, 1024);
  gemm(pA, pB, nullptr, Ls[16], 1024, 1);
  gemm(pB, f1b, nullptr, Ls[17], 1024, 1);

  // ---- FP1: interp(xyz<-l1x, f1) + [xyz^T;points] -> MLP 137-128-128-128-128 ----
  knn3_kernel<<<e1(4096), 256, 0, stream>>>(xyz, l1x, 4096, 1024, idx3, w3);
  interp_kernel<<<e1(128*4096), 256, 0, stream>>>(f1b, idx3, w3, pA, 128, 137, 1024, 4096);
  fp1unk_kernel<<<e1(9*4096), 256, 0, stream>>>(xyz, pts, pA, 4096, 137, 128);
  gemm(pA, pB, nullptr, Ls[18], 4096, 1);
  gemm(pB, pA, nullptr, Ls[19], 4096, 1);
  gemm(pA, pB, nullptr, Ls[20], 4096, 1);
  gemm(pB, l0, nullptr, Ls[21], 4096, 1);

  // ---- Head: 128-128 (leaky), 128-64 (leaky), 64-13 (linear, f32 logits) ----
  gemm(l0, pA, nullptr, Ls[22], 4096, 2);
  gemm(pA, pB, nullptr, Ls[23], 4096, 2);
  gemm(pB, nullptr, o_sem, Ls[24], 4096, 0);

  // ---- outputs ----
  transpose_pf_kernel<<<e1(128*4096), 256, 0, stream>>>(l0, o_pf, 128, 4096);
  softmax_kernel<<<B*13, 256, 0, stream>>>(o_sem, o_ys, 4096, 13);
}
